// VAEGenerator_17712445129055
// MI455X (gfx1250) — matrix operand order repaired and checked by reading, not compiled, not run on hardware
//
#include <hip/hip_runtime.h>
#include <hip/hip_bf16.h>
#include <math.h>

typedef _Float16 half_t;
typedef __attribute__((ext_vector_type(16))) _Float16 v16h;
typedef __attribute__((ext_vector_type(8)))  _Float16 v8h;
typedef __attribute__((ext_vector_type(8)))  float    v8f;

// ---------------------------------------------------------------------------
// WMMA fragment helpers (CDNA5 V_WMMA_F32_16X16X32_F16 layouts, wave32)
// ---------------------------------------------------------------------------
__device__ __forceinline__ v16h cat8(v8h lo, v8h hi) {
    v16h r;
#pragma unroll
    for (int i = 0; i < 8; ++i) { r[i] = lo[i]; r[i + 8] = hi[i]; }
    return r;
}

// A fragment (16x32, f16) from row-major f16 X (ld = K); works for global or LDS.
// lanes 0-15: row m_base+lane, K = {0..7, 16..23}; lanes 16-31: K = {8..15, 24..31}
__device__ __forceinline__ v16h load_frag_A(const half_t* X, int ld,
                                            int m_base, int kb, int lane) {
    const int hs  = lane >> 4;
    const int row = m_base + (lane & 15);
    const half_t* p = X + (size_t)row * ld + kb * 32 + hs * 8;
    v8h c0 = *(const v8h*)(p);
    v8h c1 = *(const v8h*)(p + 16);
    return cat8(c0, c1);
}

// B fragment (32x16, f16) for X @ W^T: element (k,n) = W[n][k], W row-major (N x K).
// lanes 0-15: col n_base+lane, K = {0..7, 16..23}; lanes 16-31: K = {8..15, 24..31}
__device__ __forceinline__ v16h load_frag_B(const half_t* __restrict__ W, int ld,
                                            int n_base, int kb, int lane) {
    const int hs  = lane >> 4;
    const int col = n_base + (lane & 15);
    const half_t* p = W + (size_t)col * ld + kb * 32 + hs * 8;
    v8h c0 = *(const v8h*)(p);
    v8h c1 = *(const v8h*)(p + 16);
    return cat8(c0, c1);
}

__device__ __forceinline__ v8f wmma_f16(v16h a, v16h b, v8f c) {
    return __builtin_amdgcn_wmma_f32_16x16x32_f16(false, a, false, b, (short)0, c,
                                                  false, false);
}

__device__ __forceinline__ float sigmoidf_fast(float x) {
    return 1.0f / (1.0f + __expf(-x));
}

// ---------------------------------------------------------------------------
// Weight pack: f32 (rows x src_stride, at src_off) -> f16 (rows x cols)
// ---------------------------------------------------------------------------
__global__ void pack_half_kernel(const float* __restrict__ src, half_t* __restrict__ dst,
                                 int rows, int cols, int src_stride, int src_off) {
    int idx = blockIdx.x * 256 + threadIdx.x;
    if (idx >= rows * cols) return;
    int r = idx / cols, c = idx - r * cols;
    dst[idx] = (half_t)src[(size_t)r * src_stride + src_off + c];
}

// ---------------------------------------------------------------------------
// h0 = relu(z @ W1^T + b1) @ W2^T + b2   (B=64, L=128, H=256) -- tiny, VALU
// ---------------------------------------------------------------------------
__global__ __launch_bounds__(256) void h0_kernel(const float* __restrict__ z,
                                                 const float* __restrict__ W1,
                                                 const float* __restrict__ b1,
                                                 const float* __restrict__ W2,
                                                 const float* __restrict__ b2,
                                                 float* __restrict__ h0) {
    __shared__ float t1[64 * 256];
    int tid = threadIdx.x;
    for (int o = tid; o < 64 * 256; o += 256) {
        int b = o >> 8, n = o & 255;
        float s = b1[n];
        for (int k = 0; k < 128; ++k) s += z[b * 128 + k] * W1[n * 128 + k];
        t1[o] = fmaxf(s, 0.0f);
    }
    __syncthreads();
    for (int o = tid; o < 64 * 256; o += 256) {
        int b = o >> 8, n = o & 255;
        float s = b2[n];
        for (int k = 0; k < 256; ++k) s += t1[b * 256 + k] * W2[n * 256 + k];
        h0[o] = s;
    }
}

// ---------------------------------------------------------------------------
// Embedding gather (t-major rows: r = t*64 + b), f16 output
// ---------------------------------------------------------------------------
__global__ void embed_kernel(const int* __restrict__ tokens, const float* __restrict__ emb,
                             half_t* __restrict__ x) {
    int idx = blockIdx.x * 256 + threadIdx.x;   // over 4096*256
    int r = idx >> 8, i = idx & 255;
    int t = r >> 6, b = r & 63;
    int tok = (t == 0) ? 2 : tokens[b * 64 + (t - 1)];
    x[idx] = (half_t)emb[tok * 256 + i];
}

// ---------------------------------------------------------------------------
// Generic WMMA GEMM: out = [relu](X @ W^T + bias), X f16 (M x K), W f16 (N x K)
// Block = 256 threads (8 waves), 64x64 tile per block, grid (N/64, M/64)
// ---------------------------------------------------------------------------
__global__ __launch_bounds__(256) void gemm_xwT_kernel(
    const half_t* __restrict__ X, const half_t* __restrict__ W,
    const float* __restrict__ bias, float* __restrict__ outF,
    half_t* __restrict__ outH, int N, int K, int relu) {
    const int n0_blk = blockIdx.x * 64;
    const int m0_blk = blockIdx.y * 64;
    const int wave = threadIdx.x >> 5;
    const int lane = threadIdx.x & 31;
    const int hs = lane >> 4;
#pragma unroll
    for (int rep = 0; rep < 2; ++rep) {
        const int tile = wave + rep * 8;        // 16 tiles of 16x16
        const int m0 = m0_blk + (tile >> 2) * 16;
        const int n0 = n0_blk + (tile & 3) * 16;
        v8f acc = {};
        for (int kb = 0; kb < K / 32; ++kb) {
            v16h a = load_frag_A(X, K, m0, kb, lane);
            v16h b = load_frag_B(W, K, n0, kb, lane);
            acc = wmma_f16(a, b, acc);
        }
#pragma unroll
        for (int v = 0; v < 8; ++v) {
            int row = m0 + v + hs * 8;
            int col = n0 + (lane & 15);
            float val = acc[v];
            if (bias) val += bias[col];
            if (relu) val = fmaxf(val, 0.0f);
            if (outF) outF[(size_t)row * N + col] = val;
            if (outH) outH[(size_t)row * N + col] = (half_t)val;
        }
    }
}

// ---------------------------------------------------------------------------
// Persistent 2-layer GRU recurrence. One workgroup, 16 waves.
//  - f32 master state lives in per-thread registers (each (row,jcol) element is
//    owned by exactly one thread for all 64 steps) -> full precision update.
//  - f16 mirrors of both states in LDS feed the WMMA A operands via
//    2x ds_load_b128 per fragment.
//  - gx0 precomputed (t-major, includes bih0); gx1 fused via WMMA vs Wih1.
// Wave w owns hidden-column block j = w*16..w*16+15 for all 4 row tiles.
// ---------------------------------------------------------------------------
__global__ __launch_bounds__(512) void gru_kernel(
    const float* __restrict__ h0, const float* __restrict__ gx0,
    const half_t* __restrict__ Whh0, const float* __restrict__ bhh0,
    const half_t* __restrict__ Wih1, const float* __restrict__ bih1,
    const half_t* __restrict__ Whh1, const float* __restrict__ bhh1,
    half_t* __restrict__ hidH) {
    __shared__ half_t h1h[64 * 256];   // 32 KB f16 mirror of layer-0 state
    __shared__ half_t h2h[64 * 256];   // 32 KB f16 mirror of layer-1 state
    const int tid = threadIdx.x;
    const int w = tid >> 5;            // wave id = hidden column tile
    const int lane = tid & 31;
    const int hs = lane >> 4;
    const int jcol = w * 16 + (lane & 15);

    // init: register master state + f16 mirrors
    float st1[4][8], st2[4][8];
#pragma unroll
    for (int m = 0; m < 4; ++m)
#pragma unroll
        for (int v = 0; v < 8; ++v) {
            int row = m * 16 + v + hs * 8;
            float val = h0[row * 256 + jcol];
            st1[m][v] = val;
            st2[m][v] = val;
        }
    for (int i = tid; i < 64 * 256; i += 512) {
        half_t hv = (half_t)h0[i];
        h1h[i] = hv;
        h2h[i] = hv;
    }
    __syncthreads();

#pragma unroll 1
    for (int t = 0; t < 64; ++t) {
        // prefetch next step's gx0 slab (49152 floats == 512 thr * 96 floats)
        if (t + 1 < 64)
            __builtin_prefetch(gx0 + (size_t)((t + 1) * 64) * 768 + tid * 96, 0, 1);

        // ---- layer 0: gh0 = h1 @ Whh0^T ----
#pragma unroll 1
        for (int m = 0; m < 4; ++m) {
            v8f aR = {}, aZ = {}, aN = {};
#pragma unroll
            for (int kb = 0; kb < 8; ++kb) {
                v16h a = load_frag_A(h1h, 256, m * 16, kb, lane);
                aR = wmma_f16(a, load_frag_B(Whh0, 256, w * 16, kb, lane), aR);
                aZ = wmma_f16(a, load_frag_B(Whh0, 256, 256 + w * 16, kb, lane), aZ);
                aN = wmma_f16(a, load_frag_B(Whh0, 256, 512 + w * 16, kb, lane), aN);
            }
#pragma unroll
            for (int v = 0; v < 8; ++v) {
                int row = m * 16 + v + hs * 8;                 // batch index
                const float* gp = gx0 + (size_t)(t * 64 + row) * 768;
                float xr = gp[jcol], xz = gp[256 + jcol], xn = gp[512 + jcol];
                float hr = aR[v] + bhh0[jcol];
                float hz = aZ[v] + bhh0[256 + jcol];
                float hn = aN[v] + bhh0[512 + jcol];
                float r  = sigmoidf_fast(xr + hr);
                float zg = sigmoidf_fast(xz + hz);
                float n  = tanhf(xn + r * hn);
                st1[m][v] = (1.0f - zg) * n + zg * st1[m][v];  // in-register update
            }
        }
        __syncthreads();                                        // all reads of h1h done
#pragma unroll
        for (int m = 0; m < 4; ++m)
#pragma unroll
            for (int v = 0; v < 8; ++v) {
                int row = m * 16 + v + hs * 8;
                h1h[row * 256 + jcol] = (half_t)st1[m][v];
            }
        __syncthreads();

        // ---- layer 1: gx1 = h1_new @ Wih1^T (fused), gh1 = h2 @ Whh1^T ----
#pragma unroll 1
        for (int m = 0; m < 4; ++m) {
            v8f xRr = {}, xZz = {}, xNn = {}, gR = {}, gZ = {}, gN = {};
#pragma unroll
            for (int kb = 0; kb < 8; ++kb) {
                v16h a1 = load_frag_A(h1h, 256, m * 16, kb, lane);
                v16h a2 = load_frag_A(h2h, 256, m * 16, kb, lane);
                xRr = wmma_f16(a1, load_frag_B(Wih1, 256, w * 16, kb, lane), xRr);
                xZz = wmma_f16(a1, load_frag_B(Wih1, 256, 256 + w * 16, kb, lane), xZz);
                xNn = wmma_f16(a1, load_frag_B(Wih1, 256, 512 + w * 16, kb, lane), xNn);
                gR  = wmma_f16(a2, load_frag_B(Whh1, 256, w * 16, kb, lane), gR);
                gZ  = wmma_f16(a2, load_frag_B(Whh1, 256, 256 + w * 16, kb, lane), gZ);
                gN  = wmma_f16(a2, load_frag_B(Whh1, 256, 512 + w * 16, kb, lane), gN);
            }
#pragma unroll
            for (int v = 0; v < 8; ++v) {
                float xr = xRr[v] + bih1[jcol];
                float xz = xZz[v] + bih1[256 + jcol];
                float xn = xNn[v] + bih1[512 + jcol];
                float hr = gR[v] + bhh1[jcol];
                float hz = gZ[v] + bhh1[256 + jcol];
                float hn = gN[v] + bhh1[512 + jcol];
                float r  = sigmoidf_fast(xr + hr);
                float zg = sigmoidf_fast(xz + hz);
                float n  = tanhf(xn + r * hn);
                st2[m][v] = (1.0f - zg) * n + zg * st2[m][v];
            }
        }
        __syncthreads();                                        // all reads of h2h done
#pragma unroll
        for (int m = 0; m < 4; ++m)
#pragma unroll
            for (int v = 0; v < 8; ++v) {
                int row = m * 16 + v + hs * 8;
                float val = st2[m][v];
                h2h[row * 256 + jcol] = (half_t)val;
                hidH[(size_t)(row * 64 + t) * 256 + jcol] = (half_t)val;  // (B,N,H)
            }
        __syncthreads();
    }
}

// ---------------------------------------------------------------------------
// Edge head: edge[b,i,j] = relu(hk[b,i]+hq[b,j]+b1) . w2 + b2, masked
// One block per (b,i); thread t: j = t&63, K-quarter = t>>6
// ---------------------------------------------------------------------------
__global__ __launch_bounds__(256) void edge_kernel(
    const float* __restrict__ hk, const float* __restrict__ hq,
    const float* __restrict__ b1, const float* __restrict__ w2,
    const float* __restrict__ b2, const unsigned char* __restrict__ mask,
    float* __restrict__ out) {
    const int blk = blockIdx.x;            // b*64 + i
    const int b = blk >> 6;
    const int i = blk & 63;
    __shared__ float hki[256];
    __shared__ float part[256];
    const int tid = threadIdx.x;
    hki[tid] = hk[(size_t)blk * 256 + tid];
    __syncthreads();
    const int j = tid & 63, q = tid >> 6;
    const float* hqr = hq + (size_t)(b * 64 + j) * 256 + q * 64;
    const float* b1q = b1 + q * 64;
    const float* w2q = w2 + q * 64;
    const float* hkq = hki + q * 64;
    float s = 0.0f;
#pragma unroll 4
    for (int k = 0; k < 64; ++k) {
        float v = fmaxf(hkq[k] + hqr[k] + b1q[k], 0.0f);
        s += v * w2q[k];
    }
    part[tid] = s;
    __syncthreads();
    if (tid < 64) {
        float tot = part[tid] + part[tid + 64] + part[tid + 128] + part[tid + 192] + b2[0];
        bool ok = (mask[b * 64 + i] != 0) && (mask[b * 64 + tid] != 0);
        out[(size_t)blk * 64 + tid] = ok ? tot : -INFINITY;
    }
}

// ---------------------------------------------------------------------------
// Stop head final: out[r] = s1[r,:] . w2 + b2   (4096 rows, K=128)
// ---------------------------------------------------------------------------
__global__ void stop_kernel(const float* __restrict__ s1, const float* __restrict__ w2,
                            const float* __restrict__ b2, float* __restrict__ out) {
    int r = blockIdx.x * 256 + threadIdx.x;
    float s = b2[0];
#pragma unroll 4
    for (int k = 0; k < 128; ++k) s += s1[(size_t)r * 128 + k] * w2[k];
    out[r] = s;
}

// ---------------------------------------------------------------------------
// Host launch
// ---------------------------------------------------------------------------
extern "C" void kernel_launch(void* const* d_in, const int* in_sizes, int n_in,
                              void* d_out, int out_size, void* d_ws, size_t ws_size,
                              hipStream_t stream) {
    (void)in_sizes; (void)n_in; (void)out_size; (void)ws_size;

    const float* z       = (const float*)d_in[0];
    const int*   tok     = (const int*)d_in[1];
    const unsigned char* mask = (const unsigned char*)d_in[2];
    const float* l2h_W1  = (const float*)d_in[3];
    const float* l2h_b1  = (const float*)d_in[4];
    const float* l2h_W2  = (const float*)d_in[5];
    const float* l2h_b2  = (const float*)d_in[6];
    const float* emb     = (const float*)d_in[7];
    const float* Wih0    = (const float*)d_in[8];
    const float* Whh0    = (const float*)d_in[9];
    const float* bih0    = (const float*)d_in[10];
    const float* bhh0    = (const float*)d_in[11];
    const float* Wih1    = (const float*)d_in[12];
    const float* Whh1    = (const float*)d_in[13];
    const float* bih1    = (const float*)d_in[14];
    const float* bhh1    = (const float*)d_in[15];
    const float* nh_W1   = (const float*)d_in[16];
    const float* nh_b1   = (const float*)d_in[17];
    const float* nh_W2   = (const float*)d_in[18];
    const float* nh_b2   = (const float*)d_in[19];
    const float* ek_W    = (const float*)d_in[20];
    const float* ek_b    = (const float*)d_in[21];
    const float* eq_W    = (const float*)d_in[22];
    const float* eq_b    = (const float*)d_in[23];
    const float* eh_W1   = (const float*)d_in[24];
    const float* eh_b1   = (const float*)d_in[25];
    const float* eh_W2   = (const float*)d_in[26];
    const float* eh_b2   = (const float*)d_in[27];
    const float* sh_W1   = (const float*)d_in[28];
    const float* sh_b1   = (const float*)d_in[29];
    const float* sh_W2   = (const float*)d_in[30];
    const float* sh_b2   = (const float*)d_in[31];

    float* out = (float*)d_out;
    float* out_node = out;                      // 64*64*256
    float* out_edge = out + 64 * 64 * 256;      // 64*64*64
    float* out_stop = out_edge + 64 * 64 * 64;  // 64*64

    // workspace carve-out (256-B aligned)
    char* ws = (char*)d_ws;
    size_t off = 0;
    auto carve = [&](size_t bytes) -> char* {
        off = (off + 255) & ~(size_t)255;
        char* p = ws + off;
        off += bytes;
        return p;
    };
    half_t* Wih0h  = (half_t*)carve(768 * 256 * 2);
    half_t* Whh0h  = (half_t*)carve(768 * 256 * 2);
    half_t* Wih1h  = (half_t*)carve(768 * 256 * 2);
    half_t* Whh1h  = (half_t*)carve(768 * 256 * 2);
    half_t* nhW1h  = (half_t*)carve(256 * 256 * 2);
    half_t* nhW2h  = (half_t*)carve(256 * 256 * 2);
    half_t* ekWh   = (half_t*)carve(256 * 256 * 2);
    half_t* eqWh   = (half_t*)carve(256 * 256 * 2);
    half_t* W1kh   = (half_t*)carve(256 * 256 * 2);
    half_t* W1qh   = (half_t*)carve(256 * 256 * 2);
    half_t* shW1h  = (half_t*)carve(128 * 256 * 2);
    half_t* xh     = (half_t*)carve((size_t)4096 * 256 * 2);
    float*  h0f    = (float*)carve(64 * 256 * 4);
    float*  gx0f   = (float*)carve((size_t)4096 * 768 * 4);
    half_t* hidH   = (half_t*)carve((size_t)4096 * 256 * 2);
    half_t* tnodeH = (half_t*)carve((size_t)4096 * 256 * 2);
    half_t* keysH  = (half_t*)carve((size_t)4096 * 256 * 2);
    half_t* qH     = (half_t*)carve((size_t)4096 * 256 * 2);
    float*  hkf    = (float*)carve((size_t)4096 * 256 * 4);
    float*  hqf    = (float*)carve((size_t)4096 * 256 * 4);
    float*  s1f    = (float*)carve((size_t)4096 * 128 * 4);

    auto pack = [&](const float* src, half_t* dst, int rows, int cols,
                    int stride, int coff) {
        int n = rows * cols;
        hipLaunchKernelGGL(pack_half_kernel, dim3((n + 255) / 256), dim3(256), 0,
                           stream, src, dst, rows, cols, stride, coff);
    };
    pack(Wih0, Wih0h, 768, 256, 256, 0);
    pack(Whh0, Whh0h, 768, 256, 256, 0);
    pack(Wih1, Wih1h, 768, 256, 256, 0);
    pack(Whh1, Whh1h, 768, 256, 256, 0);
    pack(nh_W1, nhW1h, 256, 256, 256, 0);
    pack(nh_W2, nhW2h, 256, 256, 256, 0);
    pack(ek_W, ekWh, 256, 256, 256, 0);
    pack(eq_W, eqWh, 256, 256, 256, 0);
    pack(eh_W1, W1kh, 256, 256, 512, 0);     // W1k = eh_W1[:, :256]
    pack(eh_W1, W1qh, 256, 256, 512, 256);   // W1q = eh_W1[:, 256:]
    pack(sh_W1, shW1h, 128, 256, 256, 0);

    hipLaunchKernelGGL(h0_kernel, dim3(1), dim3(256), 0, stream,
                       z, l2h_W1, l2h_b1, l2h_W2, l2h_b2, h0f);

    hipLaunchKernelGGL(embed_kernel, dim3(4096), dim3(256), 0, stream, tok, emb, xh);

    // gx0 = x @ Wih0^T + bih0   (M=4096 t-major, N=768, K=256)
    hipLaunchKernelGGL(gemm_xwT_kernel, dim3(12, 64), dim3(256), 0, stream,
                       xh, Wih0h, bih0, gx0f, (half_t*)nullptr, 768, 256, 0);

    hipLaunchKernelGGL(gru_kernel, dim3(1), dim3(512), 0, stream,
                       h0f, gx0f, Whh0h, bhh0, Wih1h, bih1, Whh1h, bhh1, hidH);

    // node head: t = relu(hid @ nh_W1^T + b1); node = t @ nh_W2^T + b2
    hipLaunchKernelGGL(gemm_xwT_kernel, dim3(4, 64), dim3(256), 0, stream,
                       hidH, nhW1h, nh_b1, (float*)nullptr, tnodeH, 256, 256, 1);
    hipLaunchKernelGGL(gemm_xwT_kernel, dim3(4, 64), dim3(256), 0, stream,
                       tnodeH, nhW2h, nh_b2, out_node, (half_t*)nullptr, 256, 256, 0);

    // keys / queries (f16 for next GEMM)
    hipLaunchKernelGGL(gemm_xwT_kernel, dim3(4, 64), dim3(256), 0, stream,
                       hidH, ekWh, ek_b, (float*)nullptr, keysH, 256, 256, 0);
    hipLaunchKernelGGL(gemm_xwT_kernel, dim3(4, 64), dim3(256), 0, stream,
                       hidH, eqWh, eq_b, (float*)nullptr, qH, 256, 256, 0);

    // hk = keys @ W1k^T ; hq = queries @ W1q^T   (no bias; eh_b1 added in edge)
    hipLaunchKernelGGL(gemm_xwT_kernel, dim3(4, 64), dim3(256), 0, stream,
                       keysH, W1kh, (float*)nullptr, hkf, (half_t*)nullptr, 256, 256, 0);
    hipLaunchKernelGGL(gemm_xwT_kernel, dim3(4, 64), dim3(256), 0, stream,
                       qH, W1qh, (float*)nullptr, hqf, (half_t*)nullptr, 256, 256, 0);

    // stop hidden: s1 = relu(hid @ sh_W1^T + sh_b1)  (N=128)
    hipLaunchKernelGGL(gemm_xwT_kernel, dim3(2, 64), dim3(256), 0, stream,
                       hidH, shW1h, sh_b1, s1f, (half_t*)nullptr, 128, 256, 1);

    hipLaunchKernelGGL(edge_kernel, dim3(4096), dim3(256), 0, stream,
                       hkf, hqf, eh_b1, eh_W2, eh_b2, mask, out_edge);

    hipLaunchKernelGGL(stop_kernel, dim3(16), dim3(256), 0, stream,
                       s1f, sh_W2, sh_b2, out_stop);
}
